// Loss_30348238913819
// MI455X (gfx1250) — matrix operand order repaired and checked by reading, not compiled, not run on hardware
//
#include <hip/hip_runtime.h>
#include <hip/hip_bf16.h>

typedef __bf16 bf16;
typedef __attribute__((ext_vector_type(16))) __bf16 v16bf;
typedef __attribute__((ext_vector_type(8)))  __bf16 v8bf;
typedef __attribute__((ext_vector_type(8)))  float  v8f;
typedef __attribute__((ext_vector_type(4)))  int    v4i;

#define NROWS 4096
#define DIM   512
// SMALL_NUM = ln(float32(1e-45)) = ln(2^-149) = -103.27893; arg gets /0.5 once more.
#define ARG_SMALL (-206.55786f)
// Fixed logsumexp shift: real entries are 4*cos in [-4,4]; masked entries are
// ARG_SMALL whose exp underflows to 0 under this shift exactly as it does
// under the reference's true row max.
#define LSE_SHIFT 4.0f

#if defined(__has_builtin)
#if __has_builtin(__builtin_amdgcn_global_load_async_to_lds_b128) && \
    __has_builtin(__builtin_amdgcn_s_wait_asynccnt)
#define USE_ASYNC_LDS 1
#endif
#endif

// ---------------------------------------------------------------------------
// Kernel 1: row-wise L2 normalize (f32), then split each value into bf16 hi/lo
// so 3x bf16 WMMA reproduces ~f32 GEMM accuracy.  One wave32 per row.
// ---------------------------------------------------------------------------
__global__ __launch_bounds__(256) void normalize_split_kernel(
    const float* __restrict__ zi, const float* __restrict__ z,
    bf16* __restrict__ zih, bf16* __restrict__ zil,
    bf16* __restrict__ zh,  bf16* __restrict__ zl)
{
    const int tid  = threadIdx.x;
    const int wid  = tid >> 5;
    const int lane = tid & 31;
    const int row  = blockIdx.x * 8 + wid;

    const float* src = (blockIdx.y == 0) ? zi : z;
    bf16* dh = (blockIdx.y == 0) ? zih : zh;
    bf16* dl = (blockIdx.y == 0) ? zil : zl;

    const float4* s4 = (const float4*)(src + (size_t)row * DIM);
    float4 v[4];
    float ss = 0.f;
#pragma unroll
    for (int i = 0; i < 4; ++i) {
        v[i] = s4[i * 32 + lane];
        ss += v[i].x * v[i].x + v[i].y * v[i].y + v[i].z * v[i].z + v[i].w * v[i].w;
    }
#pragma unroll
    for (int d = 1; d < 32; d <<= 1) ss += __shfl_xor(ss, d, 32);

    const float scale = 1.0f / fmaxf(__builtin_sqrtf(ss), 1e-8f);

#pragma unroll
    for (int i = 0; i < 4; ++i) {
        float vals[4] = {v[i].x, v[i].y, v[i].z, v[i].w};
        const size_t base = (size_t)row * DIM + (size_t)(i * 32 + lane) * 4;
#pragma unroll
        for (int e = 0; e < 4; ++e) {
            float xv = vals[e] * scale;
            bf16 h = (bf16)xv;                 // RNE
            bf16 l = (bf16)(xv - (float)h);    // residual
            dh[base + e] = h;
            dl[base + e] = l;
        }
    }
}

// ---------------------------------------------------------------------------
// Kernel 2: fused  cross/inter GEMM (split-bf16 WMMA) + mask logic +
// fixed-shift logsumexp + positive-loss accumulation.
// One workgroup per 16-row tile of zi_n; 8 waves each own interleaved
// 16-column tiles and sweep K=512 in steps of 32.
// ---------------------------------------------------------------------------
__global__ __launch_bounds__(256) void fused_loss_kernel(
    const bf16* __restrict__ zih, const bf16* __restrict__ zil,
    const bf16* __restrict__ zh,  const bf16* __restrict__ zl,
    const int* __restrict__ w, const int* __restrict__ y,
    float* __restrict__ acc)   // acc[0]=sum(pos*cross), acc[1]=sum(lse)
{
    __shared__ __align__(16) bf16 sAh[16 * DIM];   // A tile hi (32 KB total w/ lo)
    __shared__ __align__(16) bf16 sAl[16 * DIM];
    __shared__ float sSum[8][16];

    const int tid  = threadIdx.x;
    const int wid  = tid >> 5;
    const int lane = tid & 31;
    const int row0 = blockIdx.x * 16;

    // Stage A tile (zi rows row0..row0+15, hi & lo) into LDS via the CDNA5
    // async global->LDS path (ASYNCcnt) when available.
    {
        const uint4* gh = (const uint4*)(zih + (size_t)row0 * DIM);
        const uint4* gl = (const uint4*)(zil + (size_t)row0 * DIM);
        uint4* lh = (uint4*)sAh;
        uint4* ll = (uint4*)sAl;
#ifdef USE_ASYNC_LDS
        for (int i = tid; i < 16 * DIM / 8; i += 256) {
            __builtin_amdgcn_global_load_async_to_lds_b128(
                (__attribute__((address_space(1))) v4i*)(gh + i),
                (__attribute__((address_space(3))) v4i*)(lh + i), 0, 0);
            __builtin_amdgcn_global_load_async_to_lds_b128(
                (__attribute__((address_space(1))) v4i*)(gl + i),
                (__attribute__((address_space(3))) v4i*)(ll + i), 0, 0);
        }
        __builtin_amdgcn_s_wait_asynccnt(0);
#else
        for (int i = tid; i < 16 * DIM / 8; i += 256) { lh[i] = gh[i]; ll[i] = gl[i]; }
#endif
    }
    __syncthreads();

    // 16-bit fragment layouts used here (contiguous k, not the ISA table order; wave32):
    //  A 16x32: lanes 0-15 -> M=lane, halves = K+0..15
    //           lanes16-31 -> M=lane-16, halves = K+16..31
    //  B 32x16: lanes 0-15 -> N=lane,   halves = K+0..15
    //           lanes16-31 -> N=lane-16, halves = K+16..31
    //  C 16x16: VGPR r -> M = r + 8*(lane>=16), N = lane&15
    const int mrow  = lane & 15;
    const int aoff  = (lane < 16) ? 0 : 16;
    const int boff  = (lane < 16) ? 0 : 16;
    const int rbase = (lane >> 4) << 3;

    float s[8];
#pragma unroll
    for (int r = 0; r < 8; ++r) s[r] = 0.f;
    float posSum = 0.f;

    const bf16* arow_h = sAh + mrow * DIM;
    const bf16* arow_l = sAl + mrow * DIM;

    for (int jt = wid; jt < NROWS / 16; jt += 8) {
        const int c0 = jt * 16;
        v8f accC = {};   // cross dot accumulators
        v8f accI = {};   // inter dot accumulators

        const bf16* bzh = zh  + (size_t)(c0 + mrow) * DIM + boff;
        const bf16* bzl = zl  + (size_t)(c0 + mrow) * DIM + boff;
        const bf16* bih = zih + (size_t)(c0 + mrow) * DIM + boff;
        const bf16* bil = zil + (size_t)(c0 + mrow) * DIM + boff;

        // Prefetch this tile's mask lines; 16 WMMA k-steps hide the latency.
#pragma unroll
        for (int r = 0; r < 8; ++r) {
            const size_t moff = (size_t)(row0 + r + rbase) * NROWS + c0 + mrow;
            __builtin_prefetch(&w[moff], 0, 0);
            __builtin_prefetch(&y[moff], 0, 0);
        }

        for (int k = 0; k < DIM; k += 32) {
            // A fragments from LDS (two 16B chunks each)
            v8bf ah0 = *(const v8bf*)(arow_h + k + aoff);
            v8bf ah1 = *(const v8bf*)(arow_h + k + 8 + aoff);
            v8bf al0 = *(const v8bf*)(arow_l + k + aoff);
            v8bf al1 = *(const v8bf*)(arow_l + k + 8 + aoff);
            v16bf a_h = __builtin_shufflevector(ah0, ah1, 0,1,2,3,4,5,6,7,8,9,10,11,12,13,14,15);
            v16bf a_l = __builtin_shufflevector(al0, al1, 0,1,2,3,4,5,6,7,8,9,10,11,12,13,14,15);

            // B fragments: contiguous 32B of the (L2-resident) bf16 rows
            v16bf bz_h = *(const v16bf*)(bzh + k);
            v16bf bz_l = *(const v16bf*)(bzl + k);
            v16bf bi_h = *(const v16bf*)(bih + k);
            v16bf bi_l = *(const v16bf*)(bil + k);

            // split-bf16 ~f32 GEMM:  hi*hi + hi*lo + lo*hi
            accC = __builtin_amdgcn_wmma_f32_16x16x32_bf16(false, a_h, false, bz_h, (short)0, accC, false, false);
            accC = __builtin_amdgcn_wmma_f32_16x16x32_bf16(false, a_h, false, bz_l, (short)0, accC, false, false);
            accC = __builtin_amdgcn_wmma_f32_16x16x32_bf16(false, a_l, false, bz_h, (short)0, accC, false, false);
            accI = __builtin_amdgcn_wmma_f32_16x16x32_bf16(false, a_h, false, bi_h, (short)0, accI, false, false);
            accI = __builtin_amdgcn_wmma_f32_16x16x32_bf16(false, a_h, false, bi_l, (short)0, accI, false, false);
            accI = __builtin_amdgcn_wmma_f32_16x16x32_bf16(false, a_l, false, bi_h, (short)0, accI, false, false);
        }

        // Epilogue: exact int mask logic + positive loss + shifted sum-exp.
#pragma unroll
        for (int r = 0; r < 8; ++r) {
            const int gr = row0 + r + rbase;
            const int gc = c0 + mrow;
            const int wv = w[(size_t)gr * NROWS + gc];
            const int yv = y[(size_t)gr * NROWS + gc];
            const int w2   = wv + ((gr == gc) ? 1 : 0);
            const int posm = w2 & yv;              // in {0,1}
            const int negm = (w2 ^ 1) & (yv ^ 1);  // in {0,1}

            const float dc = accC[r];
            const float di = accI[r];
            posSum += (float)posm * (2.0f * dc);   // cross = dot / 0.5

            const float vI = (negm != 0 && di != 0.f) ? 4.0f * di : ARG_SMALL;
            const float vC = (negm != 0 && dc != 0.f) ? 4.0f * dc : ARG_SMALL;
            s[r] += __expf(vI - LSE_SHIFT) + __expf(vC - LSE_SHIFT);
        }
    }

    // Sum partial sum-exp across the 16 lanes sharing each row (xor in halves).
#pragma unroll
    for (int d = 1; d < 16; d <<= 1) {
#pragma unroll
        for (int r = 0; r < 8; ++r) s[r] += __shfl_xor(s[r], d, 32);
    }
    if ((lane & 15) == 0) {
#pragma unroll
        for (int r = 0; r < 8; ++r) sSum[wid][r + rbase] = s[r];
    }

    // Positive-loss partial: full-wave reduce, one atomic per wave.
#pragma unroll
    for (int d = 1; d < 32; d <<= 1) posSum += __shfl_xor(posSum, d, 32);
    if (lane == 0) atomicAdd(&acc[0], posSum);

    __syncthreads();

    // Wave 0 merges the 8 waves' per-row sums -> lse per row -> block sum.
    if (wid == 0) {
        float lse = 0.f;
        if (lane < 16) {
            float st = 0.f;
#pragma unroll
            for (int wq = 0; wq < 8; ++wq) st += sSum[wq][lane];
            // st==0 only if every entry of the row was masked: exact closed form.
            lse = (st > 0.f) ? (LSE_SHIFT + __logf(st))
                             : (ARG_SMALL + __logf(2.0f * (float)NROWS));
        }
#pragma unroll
        for (int d = 1; d < 32; d <<= 1) lse += __shfl_xor(lse, d, 32);
        if (lane == 0) atomicAdd(&acc[1], lse);
    }
}

// ---------------------------------------------------------------------------
__global__ void init_acc_kernel(float* acc)
{
    if (threadIdx.x < 2) acc[threadIdx.x] = 0.f;
}

__global__ void finalize_kernel(const float* __restrict__ acc, float* __restrict__ out)
{
    // loss = (-pos_sum + neg_lse_sum) / N
    out[0] = (acc[1] - acc[0]) * (1.0f / (float)NROWS);
}

// ---------------------------------------------------------------------------
extern "C" void kernel_launch(void* const* d_in, const int* in_sizes, int n_in,
                              void* d_out, int out_size, void* d_ws, size_t ws_size,
                              hipStream_t stream)
{
    const float* zi = (const float*)d_in[0];
    const float* z  = (const float*)d_in[1];
    const int*   w  = (const int*)d_in[2];
    const int*   y  = (const int*)d_in[3];
    float* out = (float*)d_out;

    char* ws = (char*)d_ws;
    const size_t asz = (size_t)NROWS * DIM * sizeof(bf16);   // 4 MiB each
    float* acc = (float*)ws;                                  // 2 floats
    bf16* zih = (bf16*)(ws + 256);
    bf16* zil = (bf16*)(ws + 256 + asz);
    bf16* zh  = (bf16*)(ws + 256 + 2 * asz);
    bf16* zl  = (bf16*)(ws + 256 + 3 * asz);                  // ~16.8 MiB total

    hipLaunchKernelGGL(init_acc_kernel, dim3(1), dim3(32), 0, stream, acc);
    hipLaunchKernelGGL(normalize_split_kernel, dim3(NROWS / 8, 2), dim3(256), 0, stream,
                       zi, z, zih, zil, zh, zl);
    hipLaunchKernelGGL(fused_loss_kernel, dim3(NROWS / 16), dim3(256), 0, stream,
                       zih, zil, zh, zl, w, y, acc);
    hipLaunchKernelGGL(finalize_kernel, dim3(1), dim3(1), 0, stream, acc, out);
}
